// PointTransformerClassifier_24343874634069
// MI455X (gfx1250) — matrix operand order repaired and checked by reading, not compiled, not run on hardware
//
#include <hip/hip_runtime.h>
#include <hip/hip_bf16.h>
#include <math.h>

// ---------------------------------------------------------------------------
// Point Transformer classifier forward for MI455X (gfx1250, wave32, WMMA).
// B=8, N=4096, D=64, K=8, L=2, NC=2.  All GEMMs use v_wmma_f32_16x16x32_f16;
// weights are staged into LDS by the Tensor Data Mover (tensor_load_to_lds)
// when the builtin is available.
// ---------------------------------------------------------------------------

typedef __attribute__((ext_vector_type(16))) _Float16 v16h;
typedef __attribute__((ext_vector_type(8)))  _Float16 v8h;
typedef __attribute__((ext_vector_type(8)))  float    v8f;
typedef __attribute__((ext_vector_type(4)))  unsigned int u32x4;
typedef __attribute__((ext_vector_type(4)))  int      i32x4;
typedef __attribute__((ext_vector_type(8)))  int      i32x8;

#define BB 8
#define NN 4096
#define DD 64
#define KK 8
#define RPT (BB * NN)          // 32768 points
#define EPSF 1e-5f

#if defined(__has_builtin)
#if __has_builtin(__builtin_amdgcn_tensor_load_to_lds)
#define HAS_TDM 1
#endif
#endif

// ===========================================================================
// WMMA GEMM:  Y[R,ND] = act( X[R,KD] @ W[KD,ND] + bias )
//   ACT 0: none   ACT 1: exact GELU   ACT 2: BN(scale,shift)+ReLU
// One 256-thread block = 8 waves; each wave computes a 16 x ND output tile.
// R is always a multiple of 128 for every call site, so there are no row
// guards anywhere (straight-line stores).
// W is DMA'd into LDS raw (f32) by the TDM, then converted to a transposed
// f16 layout ([n][k]) so both WMMA fragments are contiguous ds_load_b128
// pairs per the CDNA5 16-bit VGPR layouts.
// Block-level in-place safe (Y may alias X): a block stages all of its own
// 128 rows into LDS before any global store.
// ===========================================================================
template<int KD, int ND, int ACT>
__global__ __launch_bounds__(256) void wmma_gemm(
    const float* X, const float* __restrict__ W, const float* __restrict__ bias,
    const float* __restrict__ bng, const float* __restrict__ bnb,
    float* Y, int R)
{
  __shared__ alignas(16) _Float16 sW[ND * KD];      // transposed: sW[n*KD + k]
  __shared__ alignas(16) _Float16 sX[8][16 * KD];   // per-wave 16-row tiles

  const int tid  = threadIdx.x;
  const int w    = tid >> 5;
  const int lane = tid & 31;

#ifdef HAS_TDM
  // ---- Stage W via the Tensor Data Mover: global f32 [KD,ND] -> LDS raw ----
  __shared__ alignas(16) float sWraw[KD * ND];
  if (tid < 32) {   // one wave issues the DMA (EXEC ignored by tensor ops)
    const unsigned long long ga = (unsigned long long)(uintptr_t)W;
    const unsigned int ldsa     = (unsigned int)(uintptr_t)sWraw;
    // D# group 0: count=1, is_restore=0, lds_addr, global_addr[56:0], type=2
    u32x4 g0;
    g0[0] = 1u;
    g0[1] = ldsa;
    g0[2] = (unsigned int)ga;
    g0[3] = ((unsigned int)(ga >> 32) & 0x01FFFFFFu) | 0x80000000u;
    // D# group 1: workgroup_mask=0, data_size=2 (4 bytes),
    //   tensor_dim0=ND, tensor_dim1=KD, tile_dim0=ND, tile_dim1=KD,
    //   tensor_dim0_stride=ND.
    i32x8 g1;
    g1[0] = 0x00020000;                       // data_size = 4B
    g1[1] = (int)((unsigned)(ND & 0xFFFF) << 16);   // tensor_dim0[15:0]
    g1[2] = (int)((unsigned)KD << 16);        // dim0 hi=0 | tensor_dim1[15:0]
    g1[3] = (int)((unsigned)ND << 16);        // dim1 hi=0 | tile_dim0
    g1[4] = KD;                               // tile_dim1 | tile_dim2=0
    g1[5] = ND;                               // tensor_dim0_stride[31:0]
    g1[6] = 0;                                // stride0 hi | stride1 lo
    g1[7] = 0;
    i32x4 g2 = {0, 0, 0, 0};
    i32x4 g3 = {0, 0, 0, 0};
#if __has_include(<hip/amd_detail/amd_gfx1250_TDM.h>)
    i32x8 g4 = {0, 0, 0, 0, 0, 0, 0, 0};
    __builtin_amdgcn_tensor_load_to_lds(g0, g1, g2, g3, g4, 0);
#else
    __builtin_amdgcn_tensor_load_to_lds(g0, g1, g2, g3, 0);
#endif
    __builtin_amdgcn_s_wait_tensorcnt(0);
  }
  __syncthreads();
  // Convert + transpose in LDS: sW[n][k] = f16(sWraw[k][n])
  for (int i = tid; i < KD * ND; i += 256) {
    int k = i / ND, n = i % ND;
    sW[n * KD + k] = (_Float16)sWraw[i];
  }
#else
  // Fallback: stage W straight from global (f32 -> f16, transposed).
  for (int i = tid; i < KD * ND; i += 256) {
    int k = i / ND, n = i % ND;
    sW[n * KD + k] = (_Float16)W[i];
  }
#endif

  // Stage this wave's 16 rows of X (R % 128 == 0, no guards).
  const int rowBase = blockIdx.x * 128 + w * 16;
  for (int i = lane; i < 16 * KD; i += 32) {
    int r = i / KD, c = i % KD;
    sX[w][i] = (_Float16)X[(size_t)(rowBase + r) * KD + c];
  }
  __syncthreads();

  const int hi = lane >> 4;     // which half of the wave
  const int lo = lane & 15;

  v8f acc[ND / 16] = {};

#pragma unroll
  for (int cc = 0; cc < KD / 32; ++cc) {
    // A fragment (16x32 f16): lane holds row M=lo; element j -> K =
    // cc*32 + hi*8 + (j&7) + (j>=8 ? 16 : 0)  => two contiguous 8-half loads.
    const _Float16* ap = &sX[w][lo * KD + cc * 32 + hi * 8];
    v8h a0 = *(const v8h*)ap;
    v8h a1 = *(const v8h*)(ap + 16);
    v16h A = __builtin_shufflevector(a0, a1, 0, 1, 2, 3, 4, 5, 6, 7,
                                             8, 9, 10, 11, 12, 13, 14, 15);
#pragma unroll
    for (int t = 0; t < ND / 16; ++t) {
      // B fragment (32x16 f16): lane holds col N=t*16+lo; element j -> K =
      // cc*32 + hi*8 + (j&7) + (j>=8 ? 16 : 0)  => two contiguous 8-half loads.
      const _Float16* bp = &sW[(t * 16 + lo) * KD + cc * 32 + hi * 8];
      v8h b0 = *(const v8h*)bp;
      v8h b1 = *(const v8h*)(bp + 16);
      v16h Bf = __builtin_shufflevector(b0, b1, 0, 1, 2, 3, 4, 5, 6, 7,
                                                8, 9, 10, 11, 12, 13, 14, 15);
      acc[t] = __builtin_amdgcn_wmma_f32_16x16x32_f16(
          false, A, false, Bf, (short)0, acc[t], false, false);
    }
  }

  // Epilogue. C/D layout: lane l, vgpr r -> row M = r + hi*8, col N = lo.
  // Unguarded straight-line stores (R % 128 == 0).
  const float rs = rsqrtf(1.0f + EPSF);
#pragma unroll
  for (int t = 0; t < ND / 16; ++t) {
    const int n  = t * 16 + lo;
    const float bv = bias[n];
    const float sg = (ACT == 2) ? bng[n] * rs : 0.0f;
    const float sb = (ACT == 2) ? bnb[n]      : 0.0f;
#pragma unroll
    for (int r = 0; r < 8; ++r) {
      const int gr = rowBase + r + hi * 8;
      float val = acc[t][r] + bv;
      if (ACT == 1) {                                  // exact GELU
        val = 0.5f * val * (1.0f + erff(val * 0.70710678118654752f));
      } else if (ACT == 2) {                           // eval BN + ReLU
        val = fmaxf(sg * val + sb, 0.0f);
      }
      Y[(size_t)gr * ND + n] = val;
    }
  }
}

// ===========================================================================
// Embedding layer 1:  T0 = relu( x[R,3] @ ew1[3,64] + eb1 )
// ===========================================================================
__global__ __launch_bounds__(256) void k_mlp3(
    const float* __restrict__ X, const float* __restrict__ W,
    const float* __restrict__ b, float* __restrict__ Y)
{
  size_t t = (size_t)blockIdx.x * 256 + threadIdx.x;   // RPT*64 threads
  size_t r = t >> 6;
  int    c = (int)(t & 63);
  const float* xr = X + r * 3;
  float a = xr[0] * W[c] + xr[1] * W[64 + c] + xr[2] * W[128 + c] + b[c];
  Y[t] = fmaxf(a, 0.0f);
}

// ===========================================================================
// KNN: per query point, 8 nearest (incl. self) over its batch's 4096 points.
// LDS-tiled candidates, register-resident 8-deep sorted insert (fully
// unrolled -> stays in VGPRs).
// ===========================================================================
__global__ __launch_bounds__(256) void k_knn(
    const float* __restrict__ pos, int* __restrict__ idx)
{
  const int t = blockIdx.x * 256 + threadIdx.x;        // point id
  const int b = t >> 12;
  const int n = t & (NN - 1);
  const float* pb_ = pos + (size_t)b * NN * 3;
  const float qx = pb_[n * 3 + 0], qy = pb_[n * 3 + 1], qz = pb_[n * 3 + 2];

  float bd[KK]; int bi[KK];
#pragma unroll
  for (int j = 0; j < KK; ++j) { bd[j] = 3.0e38f; bi[j] = 0; }

  __shared__ float sp[256 * 3];
  for (int tile = 0; tile < NN; tile += 256) {
    __syncthreads();
    const int li = threadIdx.x;
    sp[li * 3 + 0] = pb_[(tile + li) * 3 + 0];
    sp[li * 3 + 1] = pb_[(tile + li) * 3 + 1];
    sp[li * 3 + 2] = pb_[(tile + li) * 3 + 2];
    if (tile + 256 < NN)
      __builtin_prefetch(&pb_[(tile + 256 + li) * 3], 0, 0);
    __syncthreads();
#pragma unroll 4
    for (int m = 0; m < 256; ++m) {
      float dx = qx - sp[m * 3 + 0];
      float dy = qy - sp[m * 3 + 1];
      float dz = qz - sp[m * 3 + 2];
      float d  = dx * dx + dy * dy + dz * dz;
      if (d < bd[KK - 1]) {
        float nd = d; int ni = tile + m;
#pragma unroll
        for (int j = 0; j < KK; ++j) {
          if (nd < bd[j]) {
            float td = bd[j]; int ti = bi[j];
            bd[j] = nd; bi[j] = ni; nd = td; ni = ti;
          }
        }
      }
    }
  }
#pragma unroll
  for (int j = 0; j < KK; ++j) idx[(size_t)t * KK + j] = bi[j];
}

// ===========================================================================
// pe layer 1:  PE[r,c] = relu( pd[r] @ pw1[3,64] + pb1 ),
//   pd = pos[b,n] - pos[b, idx[b,n,k]];  r = ((b*N+n)*K + k)
// ===========================================================================
__global__ __launch_bounds__(256) void k_pd_pe1(
    const float* __restrict__ pos, const int* __restrict__ idx,
    const float* __restrict__ W, const float* __restrict__ b,
    float* __restrict__ PE)
{
  size_t t = (size_t)blockIdx.x * 256 + threadIdx.x;   // RPT*K*64 threads
  int    c = (int)(t & 63);
  size_t r = t >> 6;                                   // (b,n,k) row
  size_t p = r >> 3;                                   // point id
  int    bb = (int)(p >> 12);
  int    nn = (int)(p & (NN - 1));
  int    j  = idx[r];
  const float* pb_ = pos + (size_t)bb * NN * 3;
  float dx = pb_[nn * 3 + 0] - pb_[j * 3 + 0];
  float dy = pb_[nn * 3 + 1] - pb_[j * 3 + 1];
  float dz = pb_[nn * 3 + 2] - pb_[j * 3 + 2];
  float a = dx * W[c] + dy * W[64 + c] + dz * W[128 + c] + b[c];
  PE[t] = fmaxf(a, 0.0f);
}

// ===========================================================================
// Vector attention: one wave32 per point, 2 channels per lane.
//   logits_k = sum_c (q_c - kg_c + pe_c) * aw_c + ab ; softmax over K=8
//   out_c    = sum_k w_k * (vg_c + pe_c)
// ===========================================================================
__global__ __launch_bounds__(256) void k_attn(
    const float* __restrict__ Q, const float* __restrict__ Kf,
    const float* __restrict__ V, const float* __restrict__ PE,
    const int* __restrict__ idx, const float* __restrict__ aw,
    const float* __restrict__ ab, float* __restrict__ out)
{
  const int lane = threadIdx.x & 31;
  const int p = blockIdx.x * 8 + (threadIdx.x >> 5);   // point id
  const int b = p >> 12;
  const float* qr = Q + (size_t)p * DD;
  const float  q0 = qr[lane], q1 = qr[lane + 32];
  const float  a0c = aw[lane], a1c = aw[lane + 32];
  const float  abv = ab[0];
  const int* ip = idx + (size_t)p * KK;

  float logits[KK];
#pragma unroll
  for (int kk = 0; kk < KK; ++kk) {
    const int j = ip[kk];
    const float* kr = Kf + ((size_t)b * NN + j) * DD;
    const float* pe = PE + ((size_t)p * KK + kk) * DD;
    float part = (q0 - kr[lane]      + pe[lane])      * a0c
               + (q1 - kr[lane + 32] + pe[lane + 32]) * a1c;
#pragma unroll
    for (int off = 16; off > 0; off >>= 1) part += __shfl_xor(part, off, 32);
    logits[kk] = part + abv;
  }
  float mx = logits[0];
#pragma unroll
  for (int kk = 1; kk < KK; ++kk) mx = fmaxf(mx, logits[kk]);
  float wgt[KK]; float den = 0.0f;
#pragma unroll
  for (int kk = 0; kk < KK; ++kk) { wgt[kk] = expf(logits[kk] - mx); den += wgt[kk]; }
  const float inv = 1.0f / den;

  float o0 = 0.0f, o1 = 0.0f;
#pragma unroll
  for (int kk = 0; kk < KK; ++kk) {
    const int j = ip[kk];
    const float* vr = V  + ((size_t)b * NN + j) * DD;
    const float* pe = PE + ((size_t)p * KK + kk) * DD;
    const float wv = wgt[kk] * inv;
    o0 += wv * (vr[lane]      + pe[lane]);
    o1 += wv * (vr[lane + 32] + pe[lane + 32]);
  }
  out[(size_t)p * DD + lane]      = o0;
  out[(size_t)p * DD + lane + 32] = o1;
}

// ===========================================================================
// Residual add + LayerNorm over D=64: one wave per point, 2 channels/lane.
//   out = g * (a+b - mean) * rsqrt(var + eps) + beta     (out may alias A/B)
// ===========================================================================
__global__ __launch_bounds__(256) void k_add_ln(
    const float* A, const float* Bv,
    const float* __restrict__ g, const float* __restrict__ be, float* out)
{
  const int lane = threadIdx.x & 31;
  const int p = blockIdx.x * 8 + (threadIdx.x >> 5);
  const size_t base = (size_t)p * DD;
  float x0 = A[base + lane]      + Bv[base + lane];
  float x1 = A[base + lane + 32] + Bv[base + lane + 32];
  float s = x0 + x1;
#pragma unroll
  for (int off = 16; off > 0; off >>= 1) s += __shfl_xor(s, off, 32);
  const float mean = s * (1.0f / 64.0f);
  const float d0 = x0 - mean, d1 = x1 - mean;
  float v = d0 * d0 + d1 * d1;
#pragma unroll
  for (int off = 16; off > 0; off >>= 1) v += __shfl_xor(v, off, 32);
  const float r = rsqrtf(v * (1.0f / 64.0f) + EPSF);
  out[base + lane]      = g[lane]      * d0 * r + be[lane];
  out[base + lane + 32] = g[lane + 32] * d1 * r + be[lane + 32];
}

// ===========================================================================
// Global max pool over N: G[b,c] = max_n Y[b,n,c]   (Y is [B,N,128])
// ===========================================================================
__global__ __launch_bounds__(256) void k_maxpool(
    const float* __restrict__ Y, float* __restrict__ G)
{
  const int t = blockIdx.x * 256 + threadIdx.x;   // 8*128 threads
  const int b = t >> 7, c = t & 127;
  const float* p = Y + (size_t)b * NN * 128 + c;
  float m = -3.0e38f;
  for (int n = 0; n < NN; ++n) m = fmaxf(m, p[(size_t)n * 128]);
  G[t] = m;
}

// ===========================================================================
// Classifier head (8 rows only): fc1(128)+BN+ReLU -> fc2(64)+BN+ReLU -> fc3(2)
// One 128-thread block per batch element.
// ===========================================================================
__global__ __launch_bounds__(128) void k_head(
    const float* __restrict__ G,
    const float* __restrict__ f1w, const float* __restrict__ f1b,
    const float* __restrict__ b1g, const float* __restrict__ b1b,
    const float* __restrict__ f2w, const float* __restrict__ f2b,
    const float* __restrict__ b2g, const float* __restrict__ b2b,
    const float* __restrict__ f3w, const float* __restrict__ f3b,
    float* __restrict__ out)
{
  const int b = blockIdx.x, c = threadIdx.x;
  __shared__ float s0[128], s1[128];
  const float rs = rsqrtf(1.0f + EPSF);
  s0[c] = G[b * 128 + c];
  __syncthreads();
  float a1 = f1b[c];
  for (int k = 0; k < 128; ++k) a1 += s0[k] * f1w[k * 128 + c];
  s1[c] = fmaxf(b1g[c] * a1 * rs + b1b[c], 0.0f);
  __syncthreads();
  if (c < 64) {
    float a2 = f2b[c];
    for (int k = 0; k < 128; ++k) a2 += s1[k] * f2w[k * 64 + c];
    s0[c] = fmaxf(b2g[c] * a2 * rs + b2b[c], 0.0f);
  }
  __syncthreads();
  if (c < 2) {
    float a3 = f3b[c];
    for (int k = 0; k < 64; ++k) a3 += s0[k] * f3w[k * 2 + c];
    out[b * 2 + c] = a3;
  }
}

// ===========================================================================
// Orchestration
// ===========================================================================
extern "C" void kernel_launch(void* const* d_in, const int* in_sizes, int n_in,
                              void* d_out, int out_size, void* d_ws, size_t ws_size,
                              hipStream_t stream) {
  const float* x    = (const float*)d_in[0];
  const float* ew1  = (const float*)d_in[1];
  const float* eb1  = (const float*)d_in[2];
  const float* ew2  = (const float*)d_in[3];
  const float* eb2  = (const float*)d_in[4];
  const float* wq   = (const float*)d_in[5];
  const float* bq   = (const float*)d_in[6];
  const float* wk   = (const float*)d_in[7];
  const float* bk   = (const float*)d_in[8];
  const float* wv   = (const float*)d_in[9];
  const float* bv   = (const float*)d_in[10];
  const float* pw1  = (const float*)d_in[11];
  const float* pb1  = (const float*)d_in[12];
  const float* pw2  = (const float*)d_in[13];
  const float* pb2  = (const float*)d_in[14];
  const float* aw   = (const float*)d_in[15];
  const float* ab   = (const float*)d_in[16];
  const float* ow   = (const float*)d_in[17];
  const float* ob   = (const float*)d_in[18];
  const float* l1g  = (const float*)d_in[19];
  const float* l1b  = (const float*)d_in[20];
  const float* l2g  = (const float*)d_in[21];
  const float* l2b  = (const float*)d_in[22];
  const float* fw1  = (const float*)d_in[23];
  const float* fb1  = (const float*)d_in[24];
  const float* fw2  = (const float*)d_in[25];
  const float* fb2  = (const float*)d_in[26];
  const float* cw   = (const float*)d_in[27];
  const float* cb   = (const float*)d_in[28];
  const float* cg   = (const float*)d_in[29];
  const float* cbe  = (const float*)d_in[30];
  const float* f1w  = (const float*)d_in[31];
  const float* f1b  = (const float*)d_in[32];
  const float* b1g  = (const float*)d_in[33];
  const float* b1b  = (const float*)d_in[34];
  const float* f2w  = (const float*)d_in[35];
  const float* f2b  = (const float*)d_in[36];
  const float* b2g  = (const float*)d_in[37];
  const float* b2b  = (const float*)d_in[38];
  const float* f3w  = (const float*)d_in[39];
  const float* f3b  = (const float*)d_in[40];
  float* out = (float*)d_out;

  // ---- workspace layout (floats) ----
  float* ws = (float*)d_ws;
  float* H   = ws;                         // [RPT,64]
  float* T0  = H  + (size_t)RPT * 64;      // [RPT,64]
  float* T1  = T0 + (size_t)RPT * 64;      // [RPT,64]
  float* Qb  = T1 + (size_t)RPT * 64;      // [RPT,64]
  float* Kb  = Qb + (size_t)RPT * 64;      // [RPT,64]
  float* Vb  = Kb + (size_t)RPT * 64;      // [RPT,64]
  float* PE  = Vb + (size_t)RPT * 64;      // [RPT*K,64]
  float* F   = PE + (size_t)RPT * KK * 64; // [RPT,128]
  float* G   = F  + (size_t)RPT * 128;     // [8,128]
  int*   IDX = (int*)(G + 1024);           // [RPT,8]

  const int gR  = RPT / 128;               // 256 blocks for 32768-row GEMMs
  const int gRK = RPT * KK / 128;          // 2048 blocks for pe GEMM

  // Embedding
  k_mlp3<<<RPT * 64 / 256, 256, 0, stream>>>(x, ew1, eb1, T0);
  wmma_gemm<64, 64, 0><<<gR, 256, 0, stream>>>(T0, ew2, eb2, nullptr, nullptr, H, RPT);

  // KNN
  k_knn<<<RPT / 256, 256, 0, stream>>>(x, IDX);

  for (int i = 0; i < 2; ++i) {
    const int wOff = i * DD * DD, vOff = i * DD;
    wmma_gemm<64, 64, 0><<<gR, 256, 0, stream>>>(H, wq + wOff, bq + vOff, nullptr, nullptr, Qb, RPT);
    wmma_gemm<64, 64, 0><<<gR, 256, 0, stream>>>(H, wk + wOff, bk + vOff, nullptr, nullptr, Kb, RPT);
    wmma_gemm<64, 64, 0><<<gR, 256, 0, stream>>>(H, wv + wOff, bv + vOff, nullptr, nullptr, Vb, RPT);

    k_pd_pe1<<<RPT * KK * 64 / 256, 256, 0, stream>>>(x, IDX, pw1 + i * 192, pb1 + vOff, PE);
    wmma_gemm<64, 64, 0><<<gRK, 256, 0, stream>>>(PE, pw2 + wOff, pb2 + vOff, nullptr, nullptr, PE, RPT * KK);

    k_attn<<<RPT / 8, 256, 0, stream>>>(Qb, Kb, Vb, PE, IDX, aw + vOff, ab + i, T0);
    wmma_gemm<64, 64, 0><<<gR, 256, 0, stream>>>(T0, ow + wOff, ob + vOff, nullptr, nullptr, T1, RPT);
    k_add_ln<<<RPT / 8, 256, 0, stream>>>(T1, H, l1g + vOff, l1b + vOff, H);

    wmma_gemm<64, 128, 1><<<gR, 256, 0, stream>>>(H, fw1 + i * 64 * 128, fb1 + i * 128, nullptr, nullptr, F, RPT);
    wmma_gemm<128, 64, 0><<<gR, 256, 0, stream>>>(F, fw2 + i * 128 * 64, fb2 + vOff, nullptr, nullptr, T1, RPT);
    k_add_ln<<<RPT / 8, 256, 0, stream>>>(H, T1, l2g + vOff, l2b + vOff, H);
  }

  // Classifier: conv(64->128) + BN + ReLU, max-pool, tiny head MLP
  wmma_gemm<64, 128, 2><<<gR, 256, 0, stream>>>(H, cw, cb, cg, cbe, F, RPT);
  k_maxpool<<<(BB * 128) / 256, 256, 0, stream>>>(F, G);
  k_head<<<BB, 128, 0, stream>>>(G, f1w, f1b, b1g, b1b, f2w, f2b, b2g, b2b, f3w, f3b, out);
}
